// GraphDecoder_4131758539406
// MI455X (gfx1250) — compile-verified, matrix operand order checked by reading, not run on hardware
//
#include <hip/hip_runtime.h>
#include <hip/hip_bf16.h>
#include <math.h>

// ---------------------------------------------------------------------------
// GraphDecoder pointer-attention, MI455X (gfx1250, wave32, WMMA).
// Fold all per-node projections onto the single query (seq len 1), leaving
// 3 bandwidth-bound streaming passes over node_embs (256MB each) plus tiny
// per-batch GEMMs on v_wmma_f32_16x16x32_bf16. Deterministic (no atomics).
// This round: batch fragment LDS loads ahead of the WMMA chain (one dscnt
// wait instead of four), compile-time K in the GEMM template.
// ---------------------------------------------------------------------------

typedef __attribute__((ext_vector_type(16))) __bf16 v16bf;
typedef __attribute__((ext_vector_type(8)))  __bf16 v8bf;
typedef __attribute__((ext_vector_type(8)))  float  v8f;
typedef __attribute__((ext_vector_type(4)))  __bf16 v4bf;

#define BATCH 256
#define NNODE 2048
#define EDIM  128
#define DDD   384   // 3*E
#define NH    8
#define HD    48
#define SPL   8     // splits per batch for streaming passes

#define RSQRT_HD 0.14433756729740643f   // 1/sqrt(48)
#define RSQRT_E  0.08838834764831845f   // 1/sqrt(128)

// ---- WMMA fragment builders (ISA 7.12.2 layouts, wave32) -------------------
// A/W-form frag 16x32 (16-bit): lane L holds row r=L&15; element j -> col
// (j<8 ? j : j+8) + (L>>4)*8  => two contiguous 8-element runs per lane.

static __device__ __forceinline__ v16bf frag_f32_fast(const float* p, int ld, int lane) {
  int r = lane & 15, hi = lane >> 4;
  const float* q0 = p + r * ld + hi * 8;
  float4 a0 = *(const float4*)(q0);
  float4 a1 = *(const float4*)(q0 + 4);
  float4 a2 = *(const float4*)(q0 + 16);
  float4 a3 = *(const float4*)(q0 + 20);
  v16bf f;
  f[0]  = (__bf16)a0.x; f[1]  = (__bf16)a0.y; f[2]  = (__bf16)a0.z; f[3]  = (__bf16)a0.w;
  f[4]  = (__bf16)a1.x; f[5]  = (__bf16)a1.y; f[6]  = (__bf16)a1.z; f[7]  = (__bf16)a1.w;
  f[8]  = (__bf16)a2.x; f[9]  = (__bf16)a2.y; f[10] = (__bf16)a2.z; f[11] = (__bf16)a2.w;
  f[12] = (__bf16)a3.x; f[13] = (__bf16)a3.y; f[14] = (__bf16)a3.z; f[15] = (__bf16)a3.w;
  return f;
}
// valid-row clamp+select (branchless); rows < 16, cols fully valid
static __device__ __forceinline__ v16bf frag_f32_fast_rows(const float* p, int ld,
                                                           int rows, int lane) {
  int r = lane & 15, hi = lane >> 4;
  int rc = r < rows ? r : 0;                 // clamp: address always in-bounds
  const float* q0 = p + rc * ld + hi * 8;
  float4 a0 = *(const float4*)(q0);
  float4 a1 = *(const float4*)(q0 + 4);
  float4 a2 = *(const float4*)(q0 + 16);
  float4 a3 = *(const float4*)(q0 + 20);
  float z = (r < rows) ? 1.0f : 0.0f;        // select via multiply
  v16bf f;
  f[0]  = (__bf16)(a0.x * z); f[1]  = (__bf16)(a0.y * z);
  f[2]  = (__bf16)(a0.z * z); f[3]  = (__bf16)(a0.w * z);
  f[4]  = (__bf16)(a1.x * z); f[5]  = (__bf16)(a1.y * z);
  f[6]  = (__bf16)(a1.z * z); f[7]  = (__bf16)(a1.w * z);
  f[8]  = (__bf16)(a2.x * z); f[9]  = (__bf16)(a2.y * z);
  f[10] = (__bf16)(a2.z * z); f[11] = (__bf16)(a2.w * z);
  f[12] = (__bf16)(a3.x * z); f[13] = (__bf16)(a3.y * z);
  f[14] = (__bf16)(a3.z * z); f[15] = (__bf16)(a3.w * z);
  return f;
}
// bf16 source (global ctx or LDS tile): 2x 16-byte loads
static __device__ __forceinline__ v16bf frag_bf16_fast(const __bf16* p, int ld, int lane) {
  int r = lane & 15, hi = lane >> 4;
  const __bf16* q0 = p + r * ld + hi * 8;    // 16B-aligned for our ld values
  v8bf lo = *(const v8bf*)(q0);
  v8bf h8 = *(const v8bf*)(q0 + 16);
  v16bf f;
#pragma unroll
  for (int j = 0; j < 8; ++j) { f[j] = lo[j]; f[j + 8] = h8[j]; }
  return f;
}
// transposed (k,n) source, fully in-bounds: 16 unguarded strided scalar loads
static __device__ __forceinline__ v16bf frag_f32T_fast(const float* p, int ld, int lane) {
  int r = lane & 15, hi = lane >> 4;
  v16bf f;
#pragma unroll
  for (int j = 0; j < 16; ++j) {
    int c = (j < 8 ? j : j + 8) + hi * 8;
    f[j] = (__bf16)p[c * ld + r];
  }
  return f;
}
// K-tail loaders: clamp index (address stays valid), select 0 (no branches)
static __device__ __forceinline__ v16bf frag_f32_tail(const float* p, int ld,
                                                      int kv, int lane) {
  int r = lane & 15, hi = lane >> 4;
  v16bf f;
#pragma unroll
  for (int j = 0; j < 16; ++j) {
    int c = (j < 8 ? j : j + 8) + hi * 8;
    int cc = c < kv ? c : 0;
    float v = p[r * ld + cc];
    f[j] = (__bf16)(c < kv ? v : 0.0f);
  }
  return f;
}
static __device__ __forceinline__ v16bf frag_bf16_tail(const __bf16* p, int ld,
                                                       int kv, int lane) {
  int r = lane & 15, hi = lane >> 4;
  v16bf f;
#pragma unroll
  for (int j = 0; j < 16; ++j) {
    int c = (j < 8 ? j : j + 8) + hi * 8;
    int cc = c < kv ? c : 0;
    __bf16 v = p[r * ld + cc];
    f[j] = c < kv ? v : (__bf16)0.0f;
  }
  return f;
}
static __device__ __forceinline__ v16bf frag_f32T_tail(const float* p, int ld,
                                                       int kv, int lane) {
  int r = lane & 15, hi = lane >> 4;
  v16bf f;
#pragma unroll
  for (int j = 0; j < 16; ++j) {
    int c = (j < 8 ? j : j + 8) + hi * 8;
    int cc = c < kv ? c : 0;
    float v = p[cc * ld + r];
    f[j] = (__bf16)(c < kv ? v : 0.0f);
  }
  return f;
}

// ---------------------------------------------------------------------------
// Pass 1a: per-(batch,split) partial sums for graph mean. Fully coalesced.
// ---------------------------------------------------------------------------
__global__ __launch_bounds__(256) void k_mean_partial(const float* __restrict__ nodes,
                                                      float* __restrict__ pm) {
  int b = blockIdx.x, s = blockIdx.y, tid = threadIdx.x;
  const float* p = nodes + ((long)b * NNODE + (long)s * 256) * EDIM;
  float acc = 0.0f;
  for (int i = 0; i < 128; ++i) acc += p[i * 256 + tid];   // contiguous 1KB/iter
  __shared__ float sm[256];
  sm[tid] = acc;
  __syncthreads();
  if (tid < 128) pm[((long)b * SPL + s) * EDIM + tid] = sm[tid] + sm[tid + 128];
}

// Pass 1b: combine partials -> ctx = [mean, first, last] in bf16 (GEMM A input)
__global__ __launch_bounds__(128) void k_ctx(const float* __restrict__ pm,
                                             const float* __restrict__ first,
                                             const float* __restrict__ last,
                                             __bf16* __restrict__ ctx) {
  int b = blockIdx.x, t = threadIdx.x;
  float g = 0.0f;
  for (int s = 0; s < SPL; ++s) g += pm[((long)b * SPL + s) * EDIM + t];
  g *= (1.0f / (float)NNODE);
  ctx[(long)b * DDD + t]        = (__bf16)g;
  ctx[(long)b * DDD + 128 + t]  = (__bf16)first[t];
  ctx[(long)b * DDD + 256 + t]  = (__bf16)last[t];
}

// ---------------------------------------------------------------------------
// Small GEMM on WMMA:  C[m,n] = sum_k A[m,k]*B(k,n) (+bias[n])
//   AB16 : A source is bf16 (else f32)
//   BW=1 : B source is W[n,k] row-major (C = A @ W^T);  BW=0 : B[k,n] row-major
//   K    : compile-time depth; tail (K%32) handled branchlessly.
// M, N multiples of 16 (true for all launches). One wave per 16x16 C tile.
// ---------------------------------------------------------------------------
template <int AB16, int BW, int K>
__global__ __launch_bounds__(32) void k_gemm_t(const float* Af, const __bf16* Ab, int lda,
                                               const float* Bw, int ldb,
                                               float* Cd, int ldc, const float* bias,
                                               int aOffZ, int bOffZ, int cOffZ, int biasOffZ) {
  int m0 = blockIdx.x * 16, n0 = blockIdx.y * 16, z = blockIdx.z;
  int lane = threadIdx.x;
  const float*  Ap  = AB16 ? nullptr : Af + (long)z * aOffZ + (long)m0 * lda;
  const __bf16* Abp = AB16 ? Ab + (long)z * aOffZ + (long)m0 * lda : nullptr;
  const float*  Bp  = Bw + (long)z * bOffZ;
  v8f acc = {0.f, 0.f, 0.f, 0.f, 0.f, 0.f, 0.f, 0.f};
  constexpr int KFULL = K & ~31;
#pragma unroll
  for (int k0 = 0; k0 < KFULL; k0 += 32) {
    v16bf a = AB16 ? frag_bf16_fast(Abp + k0, lda, lane)
                   : frag_f32_fast(Ap + k0, lda, lane);
    v16bf bb = BW ? frag_f32_fast(Bp + (long)n0 * ldb + k0, ldb, lane)
                  : frag_f32T_fast(Bp + (long)k0 * ldb + n0, ldb, lane);
    acc = __builtin_amdgcn_wmma_f32_16x16x32_bf16(false, a, false, bb,
                                                  (short)0, acc, false, false);
  }
  if constexpr (KFULL < K) {
    constexpr int kv = K - KFULL;
    v16bf a = AB16 ? frag_bf16_tail(Abp + KFULL, lda, kv, lane)
                   : frag_f32_tail(Ap + KFULL, lda, kv, lane);
    v16bf bb = BW ? frag_f32_tail(Bp + (long)n0 * ldb + KFULL, ldb, kv, lane)
                  : frag_f32T_tail(Bp + (long)KFULL * ldb + n0, ldb, kv, lane);
    acc = __builtin_amdgcn_wmma_f32_16x16x32_bf16(false, a, false, bb,
                                                  (short)0, acc, false, false);
  }
  int n = lane & 15, hi = lane >> 4;
  float* Cp = Cd + (long)z * cOffZ + n0 + n;
  float bv = bias ? (bias + (long)z * biasOffZ)[n0 + n] : 0.0f;
#pragma unroll
  for (int r = 0; r < 8; ++r) {
    int m = r + hi * 8;
    Cp[(long)(m0 + m) * ldc] = acc[r] + bv;
  }
}

// ---------------------------------------------------------------------------
// Pass 2: streaming attention. grid (B, SPL), 256 thr = 8 waves.
// Each wave owns 2 tiles of 16 nodes: stage to LDS (bf16), 4 WMMAs produce
// all 8 head logits, online softmax + weighted node accumulation in regs.
// Block-combine 8 waves, write (M,L,s[8x128]) partial per (b,split).
// ---------------------------------------------------------------------------
__global__ __launch_bounds__(256) void k_attn_partial(const float* __restrict__ nodes,
                                                      const float* __restrict__ qk,
                                                      const float* __restrict__ q,
                                                      const float* __restrict__ bk,
                                                      float* __restrict__ ap) {
  const int b = blockIdx.x, sp = blockIdx.y;
  const int tid = threadIdx.x, wave = tid >> 5, lane = tid & 31;

  // tiles: 8 * 16*136 bf16 = 34816B | logits: 8*128 f32 = 4096B | qoff 32B
  __shared__ __align__(16) unsigned char smem[34816 + 4096 + 64];
  __bf16* tile  = ((__bf16*)smem) + wave * (16 * 136);
  float*  lg    = ((float*)(smem + 34816)) + wave * 128;
  float*  qoffS = (float*)(smem + 34816 + 4096);

  if (tid < NH) {  // q_h . b_k_h, pre-scaled by 1/sqrt(HD)
    float a = 0.0f;
    const float* qr = q + (long)b * DDD + tid * HD;
    const float* br = bk + tid * HD;
    for (int j = 0; j < HD; ++j) a += qr[j] * br[j];
    qoffS[tid] = a * RSQRT_HD;
  }
  // B fragments: qk[b] is (8 heads x 128), W-form; rows 8..15 zeroed branchlessly
  const float* qkb = qk + (long)b * (NH * EDIM);
  v16bf bf0 = frag_f32_fast_rows(qkb + 0,  EDIM, NH, lane);
  v16bf bf1 = frag_f32_fast_rows(qkb + 32, EDIM, NH, lane);
  v16bf bf2 = frag_f32_fast_rows(qkb + 64, EDIM, NH, lane);
  v16bf bf3 = frag_f32_fast_rows(qkb + 96, EDIM, NH, lane);
  __syncthreads();

  float m[NH], l[NH], s[NH][4];
#pragma unroll
  for (int h = 0; h < NH; ++h) {
    m[h] = -INFINITY; l[h] = 0.0f;
#pragma unroll
    for (int j = 0; j < 4; ++j) s[h][j] = 0.0f;
  }

  for (int it = 0; it < 2; ++it) {
    const int t  = wave * 2 + it;
    const int n0 = sp * 256 + t * 16;
    const float4* src = (const float4*)(nodes + ((long)b * NNODE + n0) * EDIM);
    if (it == 0) __builtin_prefetch(src + 512, 0, 0);   // next tile
#pragma unroll
    for (int i = 0; i < 16; ++i) {                      // 8KB coalesced stage
      int fidx = lane + 32 * i;
      float4 v = src[fidx];
      int row = fidx >> 5, col = (fidx & 31) * 4;
      v4bf o;
      o[0] = (__bf16)v.x; o[1] = (__bf16)v.y; o[2] = (__bf16)v.z; o[3] = (__bf16)v.w;
      *(v4bf*)(tile + row * 136 + col) = o;
    }
    __builtin_amdgcn_wave_barrier();

    // batch all fragment loads (8x ds_load_b128, one wait), then WMMA chain
    v16bf a0 = frag_bf16_fast(tile + 0,  136, lane);
    v16bf a1 = frag_bf16_fast(tile + 32, 136, lane);
    v16bf a2 = frag_bf16_fast(tile + 64, 136, lane);
    v16bf a3 = frag_bf16_fast(tile + 96, 136, lane);
    v8f acc = {0.f, 0.f, 0.f, 0.f, 0.f, 0.f, 0.f, 0.f};
    acc = __builtin_amdgcn_wmma_f32_16x16x32_bf16(false, a0, false, bf0, (short)0, acc, false, false);
    acc = __builtin_amdgcn_wmma_f32_16x16x32_bf16(false, a1, false, bf1, (short)0, acc, false, false);
    acc = __builtin_amdgcn_wmma_f32_16x16x32_bf16(false, a2, false, bf2, (short)0, acc, false, false);
    acc = __builtin_amdgcn_wmma_f32_16x16x32_bf16(false, a3, false, bf3, (short)0, acc, false, false);

    // C layout: lane&15 = head col, rows = r + (lane>>4)*8
    {
      int cn = lane & 15, hi = lane >> 4;
      if (cn < NH) {
#pragma unroll
        for (int r = 0; r < 8; ++r)
          lg[(r + hi * 8) * NH + cn] = acc[r] * RSQRT_HD + qoffS[cn];
      }
    }
    __builtin_amdgcn_wave_barrier();

    // online softmax: pass 1 (new max + rescale), all lanes identical
#pragma unroll
    for (int h = 0; h < NH; ++h) {
      float tm = -INFINITY;
      for (int nn = 0; nn < 16; ++nn) tm = fmaxf(tm, lg[nn * NH + h]);
      float nm = fmaxf(m[h], tm);
      float sc = __expf(m[h] - nm);
      m[h] = nm; l[h] *= sc;
#pragma unroll
      for (int j = 0; j < 4; ++j) s[h][j] *= sc;
    }
    // lanes 0..15 turn logits into exp-weights in place (dedupe transcendentals)
    if (lane < 16) {
#pragma unroll
      for (int h = 0; h < NH; ++h)
        lg[lane * NH + h] = __expf(lg[lane * NH + h] - m[h]);
    }
    __builtin_amdgcn_wave_barrier();
    // accumulate s_h += w * node  (lane covers e = lane + 32j)
    for (int nn = 0; nn < 16; ++nn) {
      float x0 = (float)tile[nn * 136 + lane];
      float x1 = (float)tile[nn * 136 + lane + 32];
      float x2 = (float)tile[nn * 136 + lane + 64];
      float x3 = (float)tile[nn * 136 + lane + 96];
#pragma unroll
      for (int h = 0; h < NH; ++h) {
        float w = lg[nn * NH + h];
        l[h] += w;
        s[h][0] += w * x0; s[h][1] += w * x1; s[h][2] += w * x2; s[h][3] += w * x3;
      }
    }
  }

  // ---- block combine (overlay tile region after barrier) ----
  __syncthreads();
  float* cw = ((float*)smem) + wave * 1040;   // [h][0]=m,[h][1]=l,[h][2..129]=s
  if (lane == 0) {
#pragma unroll
    for (int h = 0; h < NH; ++h) { cw[h * 130 + 0] = m[h]; cw[h * 130 + 1] = l[h]; }
  }
#pragma unroll
  for (int h = 0; h < NH; ++h)
#pragma unroll
    for (int j = 0; j < 4; ++j)
      cw[h * 130 + 2 + lane + 32 * j] = s[h][j];
  __syncthreads();
  {
    int h = tid >> 5, ln = tid & 31;
    float M2 = -INFINITY;
    for (int w = 0; w < 8; ++w) M2 = fmaxf(M2, ((float*)smem)[w * 1040 + h * 130]);
    float L2 = 0.0f, sv[4] = {0.f, 0.f, 0.f, 0.f};
    for (int w = 0; w < 8; ++w) {
      const float* c2 = ((float*)smem) + w * 1040 + h * 130;
      float fsc = __expf(c2[0] - M2);
      L2 += c2[1] * fsc;
#pragma unroll
      for (int j = 0; j < 4; ++j) sv[j] += c2[2 + ln + 32 * j] * fsc;
    }
    float* gp = ap + (((long)b * SPL + sp) * NH + h) * 130;
    if (ln == 0) { gp[0] = M2; gp[1] = L2; }
#pragma unroll
    for (int j = 0; j < 4; ++j) gp[2 + ln + 32 * j] = sv[j];
  }
}

// combine splits -> snorm[b, h*128+e] = s/L  (fixed order, deterministic)
__global__ __launch_bounds__(256) void k_attn_combine(const float* __restrict__ ap,
                                                      float* __restrict__ sn) {
  int b = blockIdx.x, h = threadIdx.x >> 5, ln = threadIdx.x & 31;
  float M2 = -INFINITY;
  for (int s = 0; s < SPL; ++s)
    M2 = fmaxf(M2, ap[(((long)b * SPL + s) * NH + h) * 130]);
  float L2 = 0.0f, sv[4] = {0.f, 0.f, 0.f, 0.f};
  for (int s = 0; s < SPL; ++s) {
    const float* p = ap + (((long)b * SPL + s) * NH + h) * 130;
    float f = __expf(p[0] - M2);
    L2 += p[1] * f;
#pragma unroll
    for (int j = 0; j < 4; ++j) sv[j] += p[2 + ln + 32 * j] * f;
  }
  float inv = 1.0f / L2;
#pragma unroll
  for (int j = 0; j < 4; ++j)
    sn[(long)b * 1024 + h * 128 + ln + 32 * j] = sv[j] * inv;
}

// ---------------------------------------------------------------------------
// Pass 3: streaming pointer scores + argmax partials. grid (B, SPL).
// u(n) = tanh((qp . node_n)/sqrt(E)) * C ; masked -> -inf ; first-max index.
// ---------------------------------------------------------------------------
__global__ __launch_bounds__(256) void k_ptr_partial(const float* __restrict__ nodes,
                                                     const float* __restrict__ qp,
                                                     const unsigned char* __restrict__ mask,
                                                     const int* __restrict__ Cs,
                                                     float* __restrict__ ppv,
                                                     int* __restrict__ ppi) {
  int b = blockIdx.x, sp = blockIdx.y;
  int tid = threadIdx.x, wave = tid >> 5, lane = tid & 31;
  float Cf = (float)Cs[0];
  const float4 qv = ((const float4*)(qp + (long)b * EDIM))[lane];
  float best = -INFINITY; int bidx = 0x7fffffff;
  for (int i = 0; i < 32; ++i) {
    int n = sp * 256 + wave * 32 + i;
    const float4 xv = ((const float4*)(nodes + ((long)b * NNODE + n) * EDIM))[lane];
    float pd = xv.x * qv.x + xv.y * qv.y + xv.z * qv.z + xv.w * qv.w;
    pd += __shfl_xor(pd, 16); pd += __shfl_xor(pd, 8);
    pd += __shfl_xor(pd, 4);  pd += __shfl_xor(pd, 2); pd += __shfl_xor(pd, 1);
    float u = tanhf(pd * RSQRT_E) * Cf;
    if (mask[(long)b * NNODE + n]) u = -INFINITY;
    if (u > best || (u == best && n < bidx)) { best = u; bidx = n; }
  }
  __shared__ float bw[8]; __shared__ int bi[8];
  if (lane == 0) { bw[wave] = best; bi[wave] = bidx; }
  __syncthreads();
  if (tid == 0) {
    float B2 = -INFINITY; int I2 = 0x7fffffff;
    for (int w = 0; w < 8; ++w)
      if (bw[w] > B2 || (bw[w] == B2 && bi[w] < I2)) { B2 = bw[w]; I2 = bi[w]; }
    ppv[(long)b * SPL + sp] = B2; ppi[(long)b * SPL + sp] = I2;
  }
}

__global__ void k_ptr_final(const float* __restrict__ ppv, const int* __restrict__ ppi,
                            int* __restrict__ out) {
  int b = blockIdx.x;
  if (threadIdx.x == 0) {
    float B2 = -INFINITY; int I2 = 0x7fffffff;
    for (int s = 0; s < SPL; ++s) {
      float v = ppv[(long)b * SPL + s]; int i = ppi[(long)b * SPL + s];
      if (v > B2 || (v == B2 && i < I2)) { B2 = v; I2 = i; }
    }
    out[b] = (I2 == 0x7fffffff) ? 0 : I2;
  }
}

// ---------------------------------------------------------------------------
extern "C" void kernel_launch(void* const* d_in, const int* in_sizes, int n_in,
                              void* d_out, int out_size, void* d_ws, size_t ws_size,
                              hipStream_t stream) {
  const float* nodes = (const float*)d_in[0];
  const unsigned char* mask = (const unsigned char*)d_in[1];  // jnp bool = 1 byte
  const int*   Cs    = (const int*)d_in[2];
  const float* first = (const float*)d_in[3];
  const float* last  = (const float*)d_in[4];
  const float* kp_w  = (const float*)d_in[5];
  const float* q_w   = (const float*)d_in[6];
  const float* k_w   = (const float*)d_in[7];
  const float* v_w   = (const float*)d_in[8];
  const float* b_q   = (const float*)d_in[9];
  const float* b_k   = (const float*)d_in[10];
  const float* b_v   = (const float*)d_in[11];
  const float* out_w = (const float*)d_in[12];
  const float* out_b = (const float*)d_in[13];
  const float* att_w = (const float*)d_in[14];

  float* ws = (float*)d_ws;                      // ~13.6 MB total
  float*  pm   = ws;                             // 256*8*128   = 262144
  __bf16* ctx  = (__bf16*)(ws + 262144);         // 256*384 bf16 (49152 f32 slots)
  float*  q    = ws + 262144 + 49152;            // 256*384
  float*  qk   = q    + 98304;                   // 256*8*128
  float*  ap   = qk   + 262144;                  // 256*8*8*130 = 2129920
  float*  sn   = ap   + 2129920;                 // 256*1024
  float*  opre = sn   + 262144;                  // 256*384
  float*  o2   = opre + 98304;                   // 256*384
  float*  q2   = o2   + 98304;                   // 256*128
  float*  qp   = q2   + 32768;                   // 256*128
  float*  ppv  = qp   + 32768;                   // 256*8
  int*    ppi  = (int*)(ppv + 2048);             // 256*8

  // Pass 1: graph mean -> ctx (bf16)
  k_mean_partial<<<dim3(BATCH, SPL), 256, 0, stream>>>(nodes, pm);
  k_ctx<<<BATCH, 128, 0, stream>>>(pm, first, last, ctx);

  // q = ctx @ q_w^T + b_q                (M=256,N=384,K=384; A bf16, B W-form)
  k_gemm_t<1, 1, 384><<<dim3(16, 24, 1), 32, 0, stream>>>(nullptr, ctx, DDD, q_w, DDD,
                                                          q, DDD, b_q, 0, 0, 0, 0);
  // qk[b,h,:] = q_h @ k_w_h              (per-head z=8, K=48 -> tail step)
  k_gemm_t<0, 0, 48><<<dim3(16, 8, 8), 32, 0, stream>>>(q, nullptr, DDD, k_w, EDIM,
                                                        qk, NH * EDIM, nullptr,
                                                        48, 48 * EDIM, 128, 0);

  // Pass 2: streaming attention (WMMA logits + online softmax)
  k_attn_partial<<<dim3(BATCH, SPL), 256, 0, stream>>>(nodes, qk, q, b_k, ap);
  k_attn_combine<<<BATCH, 256, 0, stream>>>(ap, sn);

  // o = snorm_h @ v_w_h^T + b_v          (per-head z=8)
  k_gemm_t<0, 1, 128><<<dim3(16, 3, 8), 32, 0, stream>>>(sn, nullptr, NH * EDIM, v_w, EDIM,
                                                         opre, DDD, b_v,
                                                         128, 48 * EDIM, 48, 48);
  // o2 = o @ out_w^T + out_b
  k_gemm_t<0, 1, 384><<<dim3(16, 24, 1), 32, 0, stream>>>(opre, nullptr, DDD, out_w, DDD,
                                                          o2, DDD, out_b, 0, 0, 0, 0);
  // q2 = o2 @ att_out_w^T
  k_gemm_t<0, 1, 384><<<dim3(16, 8, 1), 32, 0, stream>>>(o2, nullptr, DDD, att_w, DDD,
                                                         q2, EDIM, nullptr, 0, 0, 0, 0);
  // qp = q2 @ kp_w                       (B normal (k,n) form)
  k_gemm_t<0, 0, 128><<<dim3(16, 8, 1), 32, 0, stream>>>(q2, nullptr, EDIM, kp_w, EDIM,
                                                         qp, EDIM, nullptr, 0, 0, 0, 0);

  // Pass 3: pointer scores + argmax
  k_ptr_partial<<<dim3(BATCH, SPL), 256, 0, stream>>>(nodes, qp, mask, Cs, ppv, ppi);
  k_ptr_final<<<BATCH, 32, 0, stream>>>(ppv, ppi, (int*)d_out);
}
